// EdgeDecoder_11192684773890
// MI455X (gfx1250) — matrix operand order repaired and checked by reading, compiled, not run on hardware
//
#include <hip/hip_runtime.h>

typedef __attribute__((ext_vector_type(16))) _Float16 v16h;
typedef __attribute__((ext_vector_type(8)))  _Float16 v8h;
typedef __attribute__((ext_vector_type(8)))  float    v8f;

#define WAVES 8
#define TPB   (WAVES * 32)
#define EPG   32            // edges per group = two 16-wide WMMA tiles

// EdgeDecoder: out[e] = relu(concat(z_start[row[e]], z_to[col[e]]) @ W1 + b1) @ W2 + b2
// H=128, 2H=256. Layer 1 computed TRANSPOSED via WMMA:
//   C(16 hidden x 16 edges) = A(W1 tile, 16x32) x B(Z^T tile, 32x16)
// Two edge tiles per iteration -> two independent WMMA accumulation chains
// sharing the same A fragments (ILP=2), 16 WMMAs per wave per barrier.
__global__ __launch_bounds__(TPB) void edge_decoder_wmma(
    const float* __restrict__ z_start,
    const float* __restrict__ z_to,
    const int*   __restrict__ row,
    const int*   __restrict__ col,
    const float* __restrict__ W1,   // [256,128] row-major
    const float* __restrict__ b1,   // [128]
    const float* __restrict__ W2,   // [128,1]
    const float* __restrict__ b2,   // [1]
    float*       __restrict__ out,  // [E]
    int E, int numGroups)
{
    __shared__ __align__(32) _Float16 sZ[EPG * 256]; // 32 edges x 256 feats (16 KB)
    __shared__ float sPartial[2][TPB];               // per-lane layer-2 partials (2 KB)

    const int tid  = threadIdx.x;
    const int wave = tid >> 5;     // owns hidden columns [16*wave, 16*wave+16)
    const int lane = tid & 31;
    const int hi   = lane >> 4;    // K-half / M-half selector
    const int ln   = lane & 15;    // A: M (hidden col); B/C: N (edge slot)

    // ---- Preload A fragments = W1 tile (f32->f16), in VGPRs for all 8 k-steps.
    // 16-bit A 16x32 layout (contiguous, matches B): M=lane%16, elem j: k=16*hi+j.
    const int colA = wave * 16 + ln;
    v16h Afrag[8];
    #pragma unroll
    for (int kk = 0; kk < 8; ++kk) {
        v16h a;
        #pragma unroll
        for (int j = 0; j < 8; ++j)
            a[j]     = (_Float16)W1[(size_t)(kk * 32 + 16 * hi + j) * 128 + colA];
        #pragma unroll
        for (int j = 0; j < 8; ++j)
            a[8 + j] = (_Float16)W1[(size_t)(kk * 32 + 16 * hi + 8 + j) * 128 + colA];
        Afrag[kk] = a;
    }

    // C layout: VGPR v, lane -> M = v + 8*hi (hidden unit), N = ln (edge).
    float b1r[8], w2r[8];
    #pragma unroll
    for (int v = 0; v < 8; ++v) {
        const int nv = wave * 16 + 8 * hi + v;
        b1r[v] = b1[nv];
        w2r[v] = W2[nv];
    }
    const float b2v = b2[0];

    // ---- Gather pipeline: each thread moves one 64B segment for TWO edge slots.
    const int el  = tid >> 4;   // edge slot base 0..15 (second slot = el+16)
    const int seg = tid & 15;   // 16-float segment of the 256-wide concat row
    float4 pA[4], pB[4];        // prefetched segments for slots el and el+16

    auto fetchseg = [&](int e, float4* dst) {
        const float* src = (seg < 8)
            ? (z_start + (size_t)row[e] * 128 + seg * 16)
            : (z_to    + (size_t)col[e] * 128 + (seg - 8) * 16);
        const float4* s4 = (const float4*)src;
        dst[0] = s4[0]; dst[1] = s4[1]; dst[2] = s4[2]; dst[3] = s4[3];
    };
    auto prefetch = [&](int g) {
        int e0 = g * EPG + el;      if (e0 >= E) e0 = E - 1;  // clamp; store guarded
        int e1 = g * EPG + el + 16; if (e1 >= E) e1 = E - 1;
        fetchseg(e0, pA);
        fetchseg(e1, pB);
    };
    auto stage = [&](const float4* p, int slot) {
        v8h h0, h1;
        h0[0]=(_Float16)p[0].x; h0[1]=(_Float16)p[0].y; h0[2]=(_Float16)p[0].z; h0[3]=(_Float16)p[0].w;
        h0[4]=(_Float16)p[1].x; h0[5]=(_Float16)p[1].y; h0[6]=(_Float16)p[1].z; h0[7]=(_Float16)p[1].w;
        h1[0]=(_Float16)p[2].x; h1[1]=(_Float16)p[2].y; h1[2]=(_Float16)p[2].z; h1[3]=(_Float16)p[2].w;
        h1[4]=(_Float16)p[3].x; h1[5]=(_Float16)p[3].y; h1[6]=(_Float16)p[3].z; h1[7]=(_Float16)p[3].w;
        _Float16* dst = &sZ[slot * 256 + seg * 16];
        *(v8h*)dst       = h0;
        *(v8h*)(dst + 8) = h1;
    };

    int g = blockIdx.x;
    if (g < numGroups) prefetch(g);

    for (; g < numGroups; g += gridDim.x) {
        __syncthreads();                 // sZ/sPartial safe to overwrite
        stage(pA, el);
        stage(pB, el + 16);
        __syncthreads();

        // Kick off next group's gather; overlaps the WMMA chains below.
        {
            int gn = g + gridDim.x;
            if (gn >= numGroups) gn = numGroups - 1;
            prefetch(gn);
        }

        // ---- Layer 1 (transposed): two independent chains of 8 WMMAs, K=32 each.
        // B 32x16 in contiguous k order (same as A): lane = N = edge, elem j -> k = 16*hi + j
        // => one contiguous 32B LDS load per k-step per tile.
        v8f C0 = {}, C1 = {};
        #pragma unroll
        for (int kk = 0; kk < 8; ++kk) {
            const v16h B0 = *(const v16h*)&sZ[ln * 256        + kk * 32 + hi * 16];
            const v16h B1 = *(const v16h*)&sZ[(ln + 16) * 256 + kk * 32 + hi * 16];
            C0 = __builtin_amdgcn_wmma_f32_16x16x32_f16(false, Afrag[kk], false, B0,
                                                        (short)0, C0, false, false);
            C1 = __builtin_amdgcn_wmma_f32_16x16x32_f16(false, Afrag[kk], false, B1,
                                                        (short)0, C1, false, false);
        }

        // ---- Layer 2: each lane holds 8 hidden units of ONE edge per tile.
        float p0 = 0.0f, p1 = 0.0f;
        #pragma unroll
        for (int v = 0; v < 8; ++v) {
            float h0 = fmaxf(C0[v] + b1r[v], 0.0f);
            float h1 = fmaxf(C1[v] + b1r[v], 0.0f);
            p0 = fmaf(h0, w2r[v], p0);
            p1 = fmaf(h1, w2r[v], p1);
        }
        sPartial[0][wave * 32 + lane] = p0;  // 16 partials/edge (8 waves x 2 halves)
        sPartial[1][wave * 32 + lane] = p1;
        __syncthreads();

        if (tid < EPG) {
            const int e = g * EPG + tid;
            if (e < E) {
                const int s = tid >> 4, t = tid & 15;
                float sum = b2v;
                #pragma unroll
                for (int w = 0; w < WAVES; ++w)
                    sum += sPartial[s][w * 32 + t] + sPartial[s][w * 32 + t + 16];
                out[e] = sum;
            }
        }
    }
}

extern "C" void kernel_launch(void* const* d_in, const int* in_sizes, int n_in,
                              void* d_out, int out_size, void* d_ws, size_t ws_size,
                              hipStream_t stream) {
    const float* z_start = (const float*)d_in[0];
    const float* z_to    = (const float*)d_in[1];
    const int*   row     = (const int*)d_in[2];
    const int*   col     = (const int*)d_in[3];
    const float* W1      = (const float*)d_in[4];
    const float* b1      = (const float*)d_in[5];
    const float* W2      = (const float*)d_in[6];
    const float* b2      = (const float*)d_in[7];
    float* out = (float*)d_out;

    const int E = in_sizes[2];
    const int numGroups = (E + EPG - 1) / EPG;
    int grid = numGroups < 2048 ? numGroups : 2048;
    if (grid < 1) grid = 1;

    edge_decoder_wmma<<<grid, TPB, 0, stream>>>(z_start, z_to, row, col,
                                                W1, b1, W2, b2, out, E, numGroups);
}
